// SelfAttentionNetwork_64244120813943
// MI455X (gfx1250) — matrix operand order repaired and checked by reading, not compiled, not run on hardware
//
#include <hip/hip_runtime.h>
#include <stdint.h>

// ---------------------------------------------------------------------------
// Problem constants (from reference): B=2, N=2048, D_MODEL=1024, ATT=1024,
// H=8, HEAD_DIM=128.  All GEMMs are bf16 WMMA (16x16x32) with fp32 accum.
// ---------------------------------------------------------------------------

typedef __attribute__((ext_vector_type(16))) __bf16 v16bf;
typedef __attribute__((ext_vector_type(8)))  float  v8f;

union ABFrag { v16bf v; uint4 u[2]; };

static __device__ __forceinline__ v8f wmma_bf16(v16bf a, v16bf b, v8f c) {
  return __builtin_amdgcn_wmma_f32_16x16x32_bf16(
      /*neg_a=*/false, a, /*neg_b=*/false, b,
      /*c_mod=*/(short)0, c, /*reuse_a=*/false, /*reuse_b=*/false);
}

static __device__ __forceinline__ float rowmax16(float x) {
  x = fmaxf(x, __shfl_xor(x, 8));
  x = fmaxf(x, __shfl_xor(x, 4));
  x = fmaxf(x, __shfl_xor(x, 2));
  x = fmaxf(x, __shfl_xor(x, 1));
  return x;
}
static __device__ __forceinline__ float rowsum16(float x) {
  x += __shfl_xor(x, 8);
  x += __shfl_xor(x, 4);
  x += __shfl_xor(x, 2);
  x += __shfl_xor(x, 1);
  return x;
}

// ---------------------------------------------------------------------------
// fp32 -> bf16 elementwise convert
// ---------------------------------------------------------------------------
__global__ void cvt_bf16_kernel(const float* __restrict__ in,
                                __bf16* __restrict__ out, int n) {
  int i = blockIdx.x * blockDim.x + threadIdx.x;
  if (i < n) out[i] = (__bf16)in[i];
}

// ---------------------------------------------------------------------------
// fp32 [rows,cols] -> bf16 transposed [cols,rows]
// ---------------------------------------------------------------------------
__global__ void tr_cvt_kernel(const float* __restrict__ in,
                              __bf16* __restrict__ out, int rows, int cols) {
  int i = blockIdx.x * blockDim.x + threadIdx.x;
  if (i < rows * cols) {
    int r = i / cols, c = i % cols;
    out[(size_t)c * rows + r] = (__bf16)in[i];
  }
}

// ---------------------------------------------------------------------------
// Pack bool mask bytes into u32 bitwords: out[(b*N+q)*(N/32)+w] bit j =
// mask[b][q][w*32+j].  (mask rows are contiguous, so word i covers bytes
// i*32 .. i*32+31 of the flat array.)
// ---------------------------------------------------------------------------
__global__ void mask_pack_kernel(const unsigned char* __restrict__ mask,
                                 uint32_t* __restrict__ out, int total_words) {
  int i = blockIdx.x * blockDim.x + threadIdx.x;
  if (i < total_words) {
    const unsigned char* p = mask + (size_t)i * 32;
    uint32_t w = 0;
#pragma unroll
    for (int j = 0; j < 32; ++j) w |= (p[j] ? 1u : 0u) << j;
    out[i] = w;
  }
}

// ---------------------------------------------------------------------------
// Generic bf16 WMMA GEMM:  C[m,n] = (sum_k A[m,k] * Bt[n,k] + bias[n]) * scl
// One wave owns a 32x64 output tile: 2 A-fragments x 4 B-fragments -> 8
// chained 16x16x32 WMMAs per 32-K step (1.5 b128 loads per WMMA).
// mode 0: fp32 out, row-major [M,N]
// mode 1: bf16 out scattered as [B,H,Nseq,HD]  (Q / K layout)
// mode 2: bf16 out scattered as [B,H,HD,Nseq]  (V-transposed layout)
// ---------------------------------------------------------------------------
__global__ __launch_bounds__(128) void gemm_bf16_kernel(
    const __bf16* __restrict__ A, const __bf16* __restrict__ Bt,
    const float* __restrict__ bias, void* __restrict__ out,
    int M, int Nn, int K, int mode, float oscale,
    int Bsz, int Nseq, int Hh, int HDim) {
  const int lane = threadIdx.x & 31;
  const int wave = threadIdx.x >> 5;
  const int tn = Nn >> 6;           // 64-wide n tiles
  const int tm = M >> 5;            // 32-tall m tiles
  int wid = blockIdx.x * (blockDim.x >> 5) + wave;
  if (wid >= tm * tn) return;
  const int mt = wid / tn;
  const int nt = wid % tn;
  const int rlo = lane & 15;        // row (A) / col (B) within 16-tile
  const int hi  = lane >> 4;        // lane half selects K sub-chunk

  const __bf16* arow0 = A + (size_t)(mt * 32 + rlo) * K;
  const __bf16* arow1 = A + (size_t)(mt * 32 + 16 + rlo) * K;

  v8f acc[2][4];
#pragma unroll
  for (int r = 0; r < 2; ++r)
#pragma unroll
    for (int j = 0; j < 4; ++j) acc[r][j] = (v8f){0,0,0,0,0,0,0,0};

  for (int k = 0; k < K; k += 32) {
    ABFrag a0, a1;
    a0.u[0] = *(const uint4*)(arow0 + k + hi * 16);
    a0.u[1] = *(const uint4*)(arow0 + k + hi * 16 + 8);
    a1.u[0] = *(const uint4*)(arow1 + k + hi * 16);
    a1.u[1] = *(const uint4*)(arow1 + k + hi * 16 + 8);
#pragma unroll
    for (int j = 0; j < 4; ++j) {
      const __bf16* bp = Bt + (size_t)(nt * 64 + j * 16 + rlo) * K + k + hi * 16;
      ABFrag b;
      b.u[0] = *(const uint4*)(bp);
      b.u[1] = *(const uint4*)(bp + 8);
      acc[0][j] = wmma_bf16(a0.v, b.v, acc[0][j]);
      acc[1][j] = wmma_bf16(a1.v, b.v, acc[1][j]);
    }
  }

#pragma unroll
  for (int r = 0; r < 2; ++r) {
#pragma unroll
    for (int j = 0; j < 4; ++j) {
      int col = nt * 64 + j * 16 + rlo;
      float bv = bias ? bias[col] : 0.0f;
#pragma unroll
      for (int v = 0; v < 8; ++v) {
        int row = mt * 32 + r * 16 + v + hi * 8;  // C-layout rows
        float val = (acc[r][j][v] + bv) * oscale;
        if (mode == 0) {
          ((float*)out)[(size_t)row * Nn + col] = val;
        } else {
          int b = row / Nseq, rr = row % Nseq;
          int h = col / HDim, d = col % HDim;
          size_t idx = (mode == 1)
              ? (((size_t)(b * Hh + h) * Nseq + rr) * HDim + d)
              : (((size_t)(b * Hh + h) * HDim + d) * Nseq + rr);
          ((__bf16*)out)[idx] = (__bf16)val;
        }
      }
    }
  }
}

// ---------------------------------------------------------------------------
// Flash-attention kernel.  One wave per (b, h, 16-row q tile).
// Q pre-scaled by d_model^-0.5 * log2(e); softmax runs in exp2 domain.
//   Kb    : [B,H,N,HD] bf16   (B-operand for S = Q * K^T)
//   Vt    : [B,H,HD,N] bf16   (B-operand for O = P * V)
//   mbits : [B,N,N/32] u32    (packed bool mask, bit=1 -> masked out)
// Output: attn [B*N, ATT] bf16 (A-operand of the final projection).
// ---------------------------------------------------------------------------
#define HD_ 128
__global__ __launch_bounds__(128) void attn_kernel(
    const __bf16* __restrict__ Qb, const __bf16* __restrict__ Kb,
    const __bf16* __restrict__ Vt, const uint32_t* __restrict__ mbits,
    __bf16* __restrict__ attn, int Bsz, int Hh, int Nn) {
  __shared__ __align__(16) __bf16 lds[4][16 * 32];   // P staging, 1KB/wave

  const int lane = threadIdx.x & 31;
  const int wave = threadIdx.x >> 5;
  const int qtiles = Nn >> 4;
  int wid = blockIdx.x * (blockDim.x >> 5) + wave;
  if (wid >= Bsz * Hh * qtiles) return;
  const int qt = wid % qtiles;
  const int bh = wid / qtiles;
  const int h  = bh % Hh;
  const int b  = bh / Hh;
  const int rlo = lane & 15;
  const int hi  = lane >> 4;
  const int mwords = Nn >> 5;

  const __bf16* Qp = Qb + (size_t)bh * Nn * HD_;
  const __bf16* Kp = Kb + (size_t)bh * Nn * HD_;
  const __bf16* Vp = Vt + (size_t)bh * HD_ * Nn;
  const uint32_t* mrow = mbits + (size_t)b * Nn * mwords;

  // ---- load the 16x128 Q tile once (4 A-fragments of 16x32) ----
  v16bf qf[4];
  const __bf16* qptr = Qp + (size_t)(qt * 16 + rlo) * HD_;
#pragma unroll
  for (int kk = 0; kk < 4; ++kk) {
    ABFrag a;
    a.u[0] = *(const uint4*)(qptr + kk * 32 + hi * 16);
    a.u[1] = *(const uint4*)(qptr + kk * 32 + hi * 16 + 8);
    qf[kk] = a.v;
  }

  v8f o[8];
  float m[8], l[8];
#pragma unroll
  for (int d = 0; d < 8; ++d) o[d] = (v8f){0,0,0,0,0,0,0,0};
#pragma unroll
  for (int v = 0; v < 8; ++v) { m[v] = -__builtin_inff(); l[v] = 0.0f; }

  for (int kc = 0; kc < Nn; kc += 32) {
    // ---- S tile: 16 q-rows x 32 keys (two 16x16 C tiles) ----
    v8f s0 = (v8f){0,0,0,0,0,0,0,0};
    v8f s1 = s0;
#pragma unroll
    for (int kk = 0; kk < 4; ++kk) {
      const __bf16* p0 = Kp + (size_t)(kc + rlo) * HD_ + kk * 32 + hi * 16;
      const __bf16* p1 = Kp + (size_t)(kc + 16 + rlo) * HD_ + kk * 32 + hi * 16;
      ABFrag b0, b1;
      b0.u[0] = *(const uint4*)(p0); b0.u[1] = *(const uint4*)(p0 + 8);
      b1.u[0] = *(const uint4*)(p1); b1.u[1] = *(const uint4*)(p1 + 8);
      s0 = wmma_bf16(qf[kk], b0.v, s0);
      s1 = wmma_bf16(qf[kk], b1.v, s1);
    }

    // ---- mask (bit tests), online softmax (exp2), stage P into LDS ----
#pragma unroll
    for (int v = 0; v < 8; ++v) {
      int row = qt * 16 + v + hi * 8;
      uint32_t mw = mrow[(size_t)row * mwords + (kc >> 5)];
      float a0 = s0[v], a1 = s1[v];
      if ((mw >> rlo) & 1u)        a0 = -__builtin_inff();
      if ((mw >> (rlo + 16)) & 1u) a1 = -__builtin_inff();
      float mc = rowmax16(fmaxf(a0, a1));
      float mn = fmaxf(m[v], mc);
      bool deg = (mn == -__builtin_inff());   // whole row masked so far
      float alpha = deg ? 1.0f : __builtin_amdgcn_exp2f(m[v] - mn);
      float p0 = deg ? 0.0f : __builtin_amdgcn_exp2f(a0 - mn);
      float p1 = deg ? 0.0f : __builtin_amdgcn_exp2f(a1 - mn);
      l[v] = l[v] * alpha + rowsum16(p0 + p1);
      m[v] = mn;
#pragma unroll
      for (int d = 0; d < 8; ++d) o[d][v] *= alpha;
      int rl = v + hi * 8;                    // local row in the 16x32 P tile
      lds[wave][rl * 32 + rlo]      = (__bf16)p0;
      lds[wave][rl * 32 + 16 + rlo] = (__bf16)p1;
    }
    asm volatile("s_wait_dscnt 0" ::: "memory");

    // ---- reload P in A-fragment layout (one 16x32 fragment) ----
    ABFrag pa;
    pa.u[0] = *(const uint4*)&lds[wave][rlo * 32 + hi * 16];
    pa.u[1] = *(const uint4*)&lds[wave][rlo * 32 + hi * 16 + 8];

    // ---- O += P * V  (8 d-tiles of 16) ----
#pragma unroll
    for (int d = 0; d < 8; ++d) {
      const __bf16* vp = Vp + (size_t)(d * 16 + rlo) * Nn + kc + hi * 16;
      ABFrag bv;
      bv.u[0] = *(const uint4*)(vp);
      bv.u[1] = *(const uint4*)(vp + 8);
      o[d] = wmma_bf16(pa.v, bv.v, o[d]);
    }
  }

  // ---- normalize and write to [B*N, ATT] bf16 (A-operand of out-proj) ----
#pragma unroll
  for (int v = 0; v < 8; ++v) {
    float inv = 1.0f / l[v];
    int grow = b * Nn + qt * 16 + v + hi * 8;
    __bf16* orow = attn + (size_t)grow * (Hh * HD_) + h * HD_;
#pragma unroll
    for (int d = 0; d < 8; ++d) {
      orow[d * 16 + rlo] = (__bf16)(o[d][v] * inv);
    }
  }
}

// ---------------------------------------------------------------------------
// Host-side launch
// ---------------------------------------------------------------------------
extern "C" void kernel_launch(void* const* d_in, const int* in_sizes, int n_in,
                              void* d_out, int out_size, void* d_ws, size_t ws_size,
                              hipStream_t stream) {
  (void)in_sizes; (void)n_in; (void)out_size; (void)ws_size;
  const int B = 2, N = 2048, DM = 1024, ATT = 1024, H = 8, HD = 128;
  const int M = B * N;                         // 4096 rows

  const float* x   = (const float*)d_in[0];
  const unsigned char* mask = (const unsigned char*)d_in[1];  // numpy bool bytes
  const float* Wq  = (const float*)d_in[2];
  const float* bq  = (const float*)d_in[3];
  const float* Wk  = (const float*)d_in[4];
  const float* bk  = (const float*)d_in[5];
  const float* Wv  = (const float*)d_in[6];
  const float* bv  = (const float*)d_in[7];
  const float* Wo  = (const float*)d_in[8];
  const float* bo  = (const float*)d_in[9];

  char* ws = (char*)d_ws;
  const size_t MiB = 1024 * 1024;
  __bf16*   xbf   = (__bf16*)(ws);              // 8 MiB  [M, DM]
  __bf16*   WqT   = (__bf16*)(ws +  8 * MiB);   // 2 MiB  [ATT, DM]
  __bf16*   WkT   = (__bf16*)(ws + 10 * MiB);
  __bf16*   WvT   = (__bf16*)(ws + 12 * MiB);
  __bf16*   WoT   = (__bf16*)(ws + 14 * MiB);   //        [DM, ATT]
  __bf16*   Qb    = (__bf16*)(ws + 16 * MiB);   // 8 MiB  [B,H,N,HD]
  __bf16*   Kb    = (__bf16*)(ws + 24 * MiB);   // 8 MiB  [B,H,N,HD]
  __bf16*   Vt    = (__bf16*)(ws + 32 * MiB);   // 8 MiB  [B,H,HD,N]
  __bf16*   attn  = (__bf16*)(ws + 40 * MiB);   // 8 MiB  [M, ATT]
  uint32_t* mbits = (uint32_t*)(ws + 48 * MiB); // 1 MiB  [B,N,N/32]

  // 1) convert x to bf16
  {
    int n = M * DM;
    cvt_bf16_kernel<<<(n + 255) / 256, 256, 0, stream>>>(x, xbf, n);
  }
  // 2) transpose+convert weights to bf16
  {
    int n = DM * ATT;
    int g = (n + 255) / 256;
    tr_cvt_kernel<<<g, 256, 0, stream>>>(Wq, WqT, DM, ATT);
    tr_cvt_kernel<<<g, 256, 0, stream>>>(Wk, WkT, DM, ATT);
    tr_cvt_kernel<<<g, 256, 0, stream>>>(Wv, WvT, DM, ATT);
    tr_cvt_kernel<<<g, 256, 0, stream>>>(Wo, WoT, ATT, DM);
  }
  // 3) pack mask bools into bitwords
  {
    int words = B * N * (N / 32);
    mask_pack_kernel<<<(words + 255) / 256, 256, 0, stream>>>(mask, mbits, words);
  }
  // 4) QKV projections (wave = 32x64 tile): waves = (M/32)*(ATT/64) = 2048
  {
    int waves = (M / 32) * (ATT / 64);
    int blocks = (waves + 3) / 4;
    // fold d_model^-0.5 * log2(e) into Q so softmax runs in exp2 domain
    const float qscale = 0.03125f * 1.44269504088896340736f;
    gemm_bf16_kernel<<<blocks, 128, 0, stream>>>(xbf, WqT, bq, Qb,
        M, ATT, DM, /*mode=*/1, qscale, B, N, H, HD);
    gemm_bf16_kernel<<<blocks, 128, 0, stream>>>(xbf, WkT, bk, Kb,
        M, ATT, DM, /*mode=*/1, 1.0f, B, N, H, HD);
    gemm_bf16_kernel<<<blocks, 128, 0, stream>>>(xbf, WvT, bv, Vt,
        M, ATT, DM, /*mode=*/2, 1.0f, B, N, H, HD);
  }
  // 5) attention: waves = B*H*(N/16) = 2048 -> 512 blocks of 4 waves
  {
    int waves = B * H * (N / 16);
    int blocks = (waves + 3) / 4;
    attn_kernel<<<blocks, 128, 0, stream>>>(Qb, Kb, Vt, mbits, attn, B, H, N);
  }
  // 6) output projection -> fp32 d_out
  {
    int waves = (M / 32) * (DM / 64);
    int blocks = (waves + 3) / 4;
    gemm_bf16_kernel<<<blocks, 128, 0, stream>>>(attn, WoT, bo, (float*)d_out,
        M, DM, ATT, /*mode=*/0, 1.0f, B, N, H, HD);
  }
}
